// MemN2N_60756607369811
// MI455X (gfx1250) — matrix operand order repaired and checked by reading, not compiled, not run on hardware
//
#include <hip/hip_runtime.h>
#include <hip/hip_bf16.h>

typedef __attribute__((ext_vector_type(16))) _Float16 v16h;
typedef __attribute__((ext_vector_type(8)))  _Float16 v8h;
typedef __attribute__((ext_vector_type(8)))  float    v8f;

#define EMBED 128
#define NQ    10
#define BATCH 16
#define BN    (BATCH * NQ)   // 160
#define SQ    20
#define SS    40
#define MEMR  50
#define VOCAB 32000

// ---------------------------------------------------------------------------
// state[bn,d] = sum_s B_w[cq[bn,s], d] * qw(s,d)
// qw(j,k) = 1 - (j+1)/J - (k+1)/d * (1 - 2(j+1)/J)
// ---------------------------------------------------------------------------
__global__ void memn2n_init_state(const int* __restrict__ cq,
                                  const float* __restrict__ B_w,
                                  float* __restrict__ state) {
    int bn = blockIdx.x;
    int d  = threadIdx.x;
    float kterm = (d + 1.0f) / (float)EMBED;
    float acc = 0.0f;
    for (int s = 0; s < SQ; ++s) {
        int tok = cq[bn * SQ + s];
        float jr = (s + 1.0f) / (float)SQ;
        float w  = 1.0f - jr - kterm * (1.0f - 2.0f * jr);
        acc += B_w[(size_t)tok * EMBED + d] * w;
    }
    state[(size_t)bn * EMBED + d] = acc;
}

// ---------------------------------------------------------------------------
// mem/outm depend only on (b, r) and are hop-invariant (tied A/C): build once.
// memb[b,r,d] = sum_s A_w[story[b,r,s], d]*sw(s,d) + TA[r,d]   (outm: C_w, TC)
// ---------------------------------------------------------------------------
__global__ void memn2n_build_mem(const int* __restrict__ story,
                                 const float* __restrict__ A_w,
                                 const float* __restrict__ C_w,
                                 const float* __restrict__ TA,
                                 const float* __restrict__ TC,
                                 float* __restrict__ memb,
                                 float* __restrict__ outm) {
    int br = blockIdx.x;           // b*MEMR + r, 0..799
    int r  = br % MEMR;
    int d  = threadIdx.x;
    float kterm = (d + 1.0f) / (float)EMBED;
    float am = 0.0f, cm = 0.0f;
    for (int s = 0; s < SS; ++s) {
        int tok = story[(size_t)br * SS + s];
        float jr = (s + 1.0f) / (float)SS;
        float w  = 1.0f - jr - kterm * (1.0f - 2.0f * jr);
        size_t off = (size_t)tok * EMBED + d;
        am += A_w[off] * w;
        cm += C_w[off] * w;
    }
    memb[(size_t)br * EMBED + d] = am + TA[r * EMBED + d];
    outm[(size_t)br * EMBED + d] = cm + TC[r * EMBED + d];
}

// ---------------------------------------------------------------------------
// Per (bn): logits[r] = <mem[b,r,:], state[bn,:]>, softmax over r,
// response[d] = sum_r p[r]*outm[b,r,d]  -> written as f16 (GEMM A operand)
// ---------------------------------------------------------------------------
__global__ void memn2n_attn(const float* __restrict__ state,
                            const float* __restrict__ memb,
                            const float* __restrict__ outm,
                            _Float16* __restrict__ resp16) {
    __shared__ float s_state[EMBED];
    __shared__ float s_logit[64];
    int bn = blockIdx.x;
    int b  = bn / NQ;
    int t  = threadIdx.x;
    s_state[t] = state[(size_t)bn * EMBED + t];
    __syncthreads();

    if (t < MEMR) {
        const float* mrow = memb + ((size_t)(b * MEMR + t)) * EMBED;
        float s = 0.0f;
        for (int k = 0; k < EMBED; ++k) s += mrow[k] * s_state[k];
        s_logit[t] = s;
    }
    __syncthreads();

    // redundant per-thread softmax reduction over 50 (trivial cost)
    float mx = -1e30f;
    for (int r = 0; r < MEMR; ++r) mx = fmaxf(mx, s_logit[r]);
    float ssum = 0.0f;
    for (int r = 0; r < MEMR; ++r) ssum += __expf(s_logit[r] - mx);
    float inv = 1.0f / ssum;

    float resp = 0.0f;
    for (int r = 0; r < MEMR; ++r) {
        float p = __expf(s_logit[r] - mx) * inv;
        resp += p * outm[((size_t)(b * MEMR + r)) * EMBED + t];
    }
    resp16[(size_t)bn * EMBED + t] = (_Float16)resp;
}

// ---------------------------------------------------------------------------
// WMMA GEMM: C[m,n] = sum_k A16[m,k] * Bt[n,k]  (+ bias[n]) (+ resid[m,n])
// A16: [M,128] f16.  Bt: [N,128] f32 row-major, converted to f16 on the fly.
// K = 128 fixed (4 x v_wmma_f32_16x16x32_f16 per tile). One 16x16 tile per
// wave; block = 4 waves = 16(M) x 64(N); grid = (N/64, M/16).
// ---------------------------------------------------------------------------
__global__ void memn2n_gemm_wmma(const _Float16* __restrict__ A16,
                                 const float* __restrict__ Bt,
                                 float* __restrict__ C,
                                 _Float16* __restrict__ C16,
                                 const float* __restrict__ bias,
                                 const float* __restrict__ resid,
                                 int N) {
    int wave = threadIdx.x >> 5;
    int lane = threadIdx.x & 31;
    int h = lane >> 4;       // half-wave select
    int l = lane & 15;
    int n0 = blockIdx.x * 64 + wave * 16;
    int m0 = blockIdx.y * 16;

    const _Float16* Arow = A16 + (size_t)(m0 + l) * EMBED;
    const float*    Brow = Bt  + (size_t)(n0 + l) * EMBED + 16 * h;

    v8f acc = {};
#pragma unroll
    for (int t = 0; t < 4; ++t) {
        int kb = t * 32;
        // A 16x32 f16 tile: lane = M row; half 0 -> K {kb..kb+15},
        //                                 half 1 -> K {kb+16..kb+31}
        v8h alo = *(const v8h*)(Arow + kb + 16 * h);
        v8h ahi = *(const v8h*)(Arow + kb + 16 * h + 8);
        v16h a;
#pragma unroll
        for (int i = 0; i < 8; ++i) { a[i] = alo[i]; a[8 + i] = ahi[i]; }

        // B 32x16 tile: lane = N column; half h holds K kb+16h .. kb+16h+15
        const float4* B4 = (const float4*)(Brow + kb);
        v16h bfrag;
#pragma unroll
        for (int j = 0; j < 4; ++j) {
            float4 v = B4[j];
            bfrag[4 * j + 0] = (_Float16)v.x;
            bfrag[4 * j + 1] = (_Float16)v.y;
            bfrag[4 * j + 2] = (_Float16)v.z;
            bfrag[4 * j + 3] = (_Float16)v.w;
        }
        acc = __builtin_amdgcn_wmma_f32_16x16x32_f16(
            false, a, false, bfrag, (short)0, acc, false, false);
    }

    // C/D layout: lanes 0-15 -> M=j, lanes 16-31 -> M=j+8; N = l
    int n = n0 + l;
#pragma unroll
    for (int j = 0; j < 8; ++j) {
        int m = m0 + j + 8 * h;
        float v = acc[j];
        if (bias)  v += bias[n];
        if (resid) v += resid[(size_t)m * N + n];   // same lane wrote/reads elem: in-place safe
        C[(size_t)m * N + n] = v;
        if (C16) C16[(size_t)m * N + n] = (_Float16)v;
    }
}

// ---------------------------------------------------------------------------
extern "C" void kernel_launch(void* const* d_in, const int* in_sizes, int n_in,
                              void* d_out, int out_size, void* d_ws, size_t ws_size,
                              hipStream_t stream) {
    const int*   cq    = (const int*)d_in[0];   // [16,10,20]
    const int*   story = (const int*)d_in[1];   // [16,50,40]
    const float* A_w   = (const float*)d_in[2]; // [32000,128]
    const float* C_w   = (const float*)d_in[3];
    const float* B_w   = (const float*)d_in[4];
    const float* H_w   = (const float*)d_in[5]; // [128,128]
    const float* H_b   = (const float*)d_in[6]; // [128]
    const float* out_w = (const float*)d_in[7]; // [32000,128]
    const float* TA    = (const float*)d_in[8]; // [50,128]
    const float* TC    = (const float*)d_in[9]; // [50,128]
    float* out = (float*)d_out;                 // [160,32000]

    // workspace layout (~0.96 MB, all 256B-aligned)
    float*    state   = (float*)d_ws;                       // 160*128 f32
    float*    memb    = state + BN * EMBED;                 // 800*128 f32
    float*    outm    = memb + BATCH * MEMR * EMBED;        // 800*128 f32
    _Float16* resp16  = (_Float16*)(outm + BATCH * MEMR * EMBED); // 160*128 f16
    _Float16* state16 = resp16 + BN * EMBED;                // 160*128 f16

    memn2n_init_state<<<BN, EMBED, 0, stream>>>(cq, B_w, state);
    memn2n_build_mem<<<BATCH * MEMR, EMBED, 0, stream>>>(story, A_w, C_w, TA, TC,
                                                         memb, outm);
    for (int hop = 0; hop < 3; ++hop) {
        memn2n_attn<<<BN, EMBED, 0, stream>>>(state, memb, outm, resp16);
        // state = resp @ H_w^T + H_b + state  (in-place residual), also emit f16
        memn2n_gemm_wmma<<<dim3(EMBED / 64, BN / 16), 128, 0, stream>>>(
            resp16, H_w, state, state16, H_b, state, EMBED);
    }
    // logits = state @ out_w^T
    memn2n_gemm_wmma<<<dim3(VOCAB / 64, BN / 16), 128, 0, stream>>>(
        state16, out_w, out, nullptr, nullptr, nullptr, VOCAB);
}
